// HOPEBlock_75393855914508
// MI455X (gfx1250) — compile-verified, matrix operand order checked by reading, not run on hardware
//
#include <hip/hip_runtime.h>
#include <cstdint>
#include <cstddef>

// ---------------------------------------------------------------------------
// Types for CDNA5 WMMA (wave32, 16x16x32 bf16 -> f32) and TDM descriptors
// ---------------------------------------------------------------------------
typedef __attribute__((ext_vector_type(16))) __bf16 v16bf;
typedef __attribute__((ext_vector_type(8)))  __bf16 v8bf;
typedef __attribute__((ext_vector_type(8)))  float  v8f;
typedef __attribute__((ext_vector_type(4)))  unsigned int u32x4;
typedef __attribute__((ext_vector_type(8)))  int i32x8;
typedef __attribute__((ext_vector_type(4)))  int i32x4;

#define DIMV     512
#define HIDV     2048
#define NTOK     8192        // B*S = 4*2048
#define SEQ      2048
#define NH       8
#define HD       64
#define NB       4

#if __has_builtin(__builtin_amdgcn_tensor_load_to_lds) && \
    __has_builtin(__builtin_amdgcn_s_wait_tensorcnt)
#define USE_TDM 1
#else
#define USE_TDM 0
#endif

__device__ __forceinline__ v8f zero8() {
    v8f r;
#pragma unroll
    for (int i = 0; i < 8; ++i) r[i] = 0.0f;
    return r;
}

// Build a 16-element bf16 fragment for one lane from a row pointer.
// ISA 7.12.2 (16-bit A/B layout): lanes 0-15 hold K {0..7,16..23},
// lanes 16-31 hold K {8..15,24..31} of the 32-wide k-step.
__device__ __forceinline__ v16bf load_frag(const __bf16* rowp, int kb, int lane) {
    const int off = (lane & 16) ? 8 : 0;
    const __bf16* p = rowp + kb + off;
    v8bf lo = *(const v8bf*)(p);
    v8bf hi = *(const v8bf*)(p + 16);
    v16bf r;
#pragma unroll
    for (int i = 0; i < 8; ++i) { r[i] = lo[i]; r[i + 8] = hi[i]; }
    return r;
}

// Same fragment but source rows are f32 (LDS-resident M state), convert on read.
__device__ __forceinline__ v16bf load_frag_f32(const float* rowp, int kb, int lane) {
    const int off = (lane & 16) ? 8 : 0;
    const float* p = rowp + kb + off;
    v16bf r;
#pragma unroll
    for (int i = 0; i < 8; ++i) { r[i] = (__bf16)p[i]; r[i + 8] = (__bf16)p[i + 16]; }
    return r;
}

__device__ __forceinline__ float gelu_tanh(float x) {
    const float x3 = x * x * x;
    return 0.5f * x * (1.0f + tanhf(0.7978845608028654f * (x + 0.044715f * x3)));
}

__device__ __forceinline__ v8f wmma_bf16(v16bf a, v16bf b, v8f c) {
    return __builtin_amdgcn_wmma_f32_16x16x32_bf16(false, a, false, b, (short)0, c,
                                                   false, false);
}

#if USE_TDM
// Issue a TDM 2-D tile load: bf16 tile [tile_y rows][tile_x elems], row stride
// `stride_elems` (in elements), global->LDS. Descriptor packing per ISA 8.3/8.4.
// 6-arg builtin form (clang-23 / therock headers): groups 0..3 + extra group + cpol.
__device__ __forceinline__ void tdm_load_2d(const void* lds_dst, const void* gsrc,
                                            unsigned tile_x, unsigned tile_y,
                                            unsigned stride_elems) {
    const unsigned long long ga = (unsigned long long)(uintptr_t)gsrc;
    const unsigned lds_addr = (unsigned)(uintptr_t)lds_dst;   // low 32 bits = LDS offset
    u32x4 g0;
    g0[0] = 1u;                                               // count=1, user descriptor
    g0[1] = lds_addr;                                         // lds_addr [63:32]
    g0[2] = (unsigned)(ga & 0xFFFFFFFFull);                   // global_addr lo
    g0[3] = (unsigned)((ga >> 32) & 0x1FFFFFFull) | (2u << 30); // addr hi | type=2
    i32x8 g1;
    g1[0] = 1 << 16;                         // wg_mask=0, data_size=1 (2 bytes)
    g1[1] = (int)(tile_x << 16);             // tensor_dim0[15:0]  (= tile_x)
    g1[2] = (int)(tile_y << 16);             // tensor_dim0 hi=0, tensor_dim1[15:0]
    g1[3] = (int)(tile_x << 16);             // tensor_dim1 hi=0, tile_dim0
    g1[4] = (int)(tile_y & 0xFFFFu);         // tile_dim1, tile_dim2=0
    g1[5] = (int)stride_elems;               // tensor_dim0_stride[31:0] (elem units)
    g1[6] = 0;                               // stride hi, dim1_stride lo
    g1[7] = 0;
    const i32x4 z4 = {0, 0, 0, 0};           // groups 2/3: higher dims unused
    const i32x8 z8 = {0, 0, 0, 0, 0, 0, 0, 0};
    __builtin_amdgcn_tensor_load_to_lds(g0, g1, z4, z4, z8, 0);
}
#endif

// ---------------------------------------------------------------------------
// Weight prep: f32 -> bf16 convert, and transpose+convert ([R][C] -> [C][R])
// ---------------------------------------------------------------------------
__global__ void cvt_bf16_kernel(const float* __restrict__ in, __bf16* __restrict__ out, int n) {
    int i = blockIdx.x * 256 + threadIdx.x;
    if (i < n) out[i] = (__bf16)in[i];
}

__global__ void transpose_cvt_kernel(const float* __restrict__ in, __bf16* __restrict__ out,
                                     int R, int C) {
    int i = blockIdx.x * 256 + threadIdx.x;
    if (i < R * C) {
        int r = i / C, c = i - r * C;
        out[(size_t)c * R + r] = (__bf16)in[i];
    }
}

// ---------------------------------------------------------------------------
// LayerNorm (per 512-elem row) -> bf16
// ---------------------------------------------------------------------------
__global__ __launch_bounds__(256) void ln_bf16_kernel(const float* __restrict__ x,
                                                      const float* __restrict__ w,
                                                      const float* __restrict__ b,
                                                      __bf16* __restrict__ out) {
    __shared__ float red[256];
    const int tid = threadIdx.x;
    const size_t row = blockIdx.x;
    const float* xr = x + row * DIMV;
    float v0 = xr[tid], v1 = xr[tid + 256];

    red[tid] = v0 + v1;
    __syncthreads();
#pragma unroll
    for (int off = 128; off > 0; off >>= 1) {
        if (tid < off) red[tid] += red[tid + off];
        __syncthreads();
    }
    const float mean = red[0] * (1.0f / DIMV);
    __syncthreads();
    const float d0 = v0 - mean, d1 = v1 - mean;
    red[tid] = d0 * d0 + d1 * d1;
    __syncthreads();
#pragma unroll
    for (int off = 128; off > 0; off >>= 1) {
        if (tid < off) red[tid] += red[tid + off];
        __syncthreads();
    }
    const float rstd = rsqrtf(red[0] * (1.0f / DIMV) + 1e-5f);
    out[row * DIMV + tid]       = (__bf16)(d0 * rstd * w[tid] + b[tid]);
    out[row * DIMV + tid + 256] = (__bf16)(d1 * rstd * w[tid + 256] + b[tid + 256]);
}

// ---------------------------------------------------------------------------
// Tiled WMMA GEMM: C[M][N] = A[M][K](bf16) * Bt[N][K]^T (+bias) (gelu) (+res)
// Block: 256 threads = 8 waves; block tile 128 rows x 64 cols; wave = 16x64.
// B slab (64 rows x 128 k-elems) staged in LDS via TDM, double buffered:
// wave0 issues tensor_load_to_lds for chunk kc+1 while all waves run WMMAs
// on chunk kc; completion enforced with s_wait_tensorcnt (in-order TDM).
// ---------------------------------------------------------------------------
template <int K, int N, bool HAS_BIAS, bool GELU, bool HAS_RES, bool OUT_BF16>
__global__ __launch_bounds__(256) void gemm_wmma_kernel(const __bf16* __restrict__ A,
                                                        const __bf16* __restrict__ Bt,
                                                        const float* __restrict__ bias,
                                                        const float* __restrict__ res,
                                                        void* __restrict__ out) {
    constexpr int KC = 128;             // k-chunk staged per TDM transfer
    constexpr int NKC = K / KC;
    const int lane = threadIdx.x & 31;
    const int wave = threadIdx.x >> 5;         // 0..7
    const int ln15 = lane & 15;
    const int rowA = blockIdx.y * 128 + wave * 16 + ln15;
    const int colBase = blockIdx.x * 64;

    __shared__ __align__(16) __bf16 Bs[2][64 * KC];

    v8f acc[4] = {zero8(), zero8(), zero8(), zero8()};
    const __bf16* arow = A + (size_t)rowA * K;
    const __bf16* bslab = Bt + (size_t)colBase * K;

#if USE_TDM
    if (wave == 0) tdm_load_2d(&Bs[0][0], bslab, KC, 64, K);
#endif

    for (int kc = 0; kc < NKC; ++kc) {
        __syncthreads();   // all readers of buffer (kc+1)&1 (chunk kc-1) are done
#if USE_TDM
        if (wave == 0) {
            if (kc + 1 < NKC) {
                tdm_load_2d(&Bs[(kc + 1) & 1][0], bslab + (size_t)(kc + 1) * KC, KC, 64, K);
                __builtin_amdgcn_s_wait_tensorcnt(1);   // chunk kc complete (in-order)
            } else {
                __builtin_amdgcn_s_wait_tensorcnt(0);
            }
        }
#else
        {   // fallback: cooperative vector loads of the B slab chunk
            for (int i = threadIdx.x; i < 64 * KC / 8; i += 256) {
                const int r = (i * 8) / KC, c = (i * 8) % KC;
                *(v8bf*)&Bs[kc & 1][r * KC + c] =
                    *(const v8bf*)&bslab[(size_t)r * K + kc * KC + c];
            }
        }
#endif
        __syncthreads();   // chunk kc visible to every wave

        const __bf16* bbuf = &Bs[kc & 1][0];
#pragma unroll
        for (int kb2 = 0; kb2 < KC; kb2 += 32) {
            const int kb = kc * KC + kb2;
            if (kb + 32 < K) __builtin_prefetch(arow + kb + 32, 0, 1);
            v16bf af = load_frag(arow, kb, lane);
#pragma unroll
            for (int j = 0; j < 4; ++j) {
                v16bf bf = load_frag(bbuf + (size_t)(j * 16 + ln15) * KC, kb2, lane);
                acc[j] = wmma_bf16(af, bf, acc[j]);
            }
        }
    }

    const int mrow0 = blockIdx.y * 128 + wave * 16 + ((lane >> 4) << 3);
#pragma unroll
    for (int j = 0; j < 4; ++j) {
        const int col = colBase + j * 16 + ln15;
#pragma unroll
        for (int r = 0; r < 8; ++r) {
            const size_t idx = (size_t)(mrow0 + r) * N + col;
            float v = acc[j][r];
            if (HAS_BIAS) v += bias[col];
            if (GELU)     v = gelu_tanh(v);
            if (HAS_RES)  v += res[idx];
            if (OUT_BF16) ((__bf16*)out)[idx] = (__bf16)v;
            else          ((float*)out)[idx]  = v;
        }
    }
}

// ---------------------------------------------------------------------------
// Self-modulating attention memory scan (chunkwise linear attention).
// One block per (b,h); 512 threads = 16 waves; wave(ti,tj) owns a 16x16 tile
// of every 64x64 GEMM. M state (64x64 f32) lives in LDS across chunks.
//   y[s] = Q[s]·M_prevᵀ + (1/B)·Σ_b' mask(Q Kᵀ_{b'}) V_{b'}
//   M   += (1/B)·Σ_b' V_{b'}ᵀ K_{b'}        (inclusive cumsum)
// K/V 64x64 tiles are fetched by the Tensor Data Mover (2-D strided tile,
// row stride 512 elems); wave 0 issues + waits TENSORcnt, block barriers.
// ---------------------------------------------------------------------------
__global__ __launch_bounds__(512) void attn_mem_kernel(const __bf16* __restrict__ Qb,
                                                       const __bf16* __restrict__ Kb,
                                                       const __bf16* __restrict__ Vb,
                                                       __bf16* __restrict__ Yb) {
    const int bh = blockIdx.x;
    const int b = bh >> 3;
    const int h = bh & 7;
    const int tid = threadIdx.x;
    const int lane = tid & 31;
    const int ln15 = lane & 15;
    const int wv = tid >> 5;          // 0..15
    const int ti = wv >> 2;           // row tile
    const int tj = wv & 3;            // col tile
    const int hcol = h * HD;

    __shared__ __align__(16) __bf16 Qs[64 * 64];   // [s][k]
    __shared__ __align__(16) __bf16 Ks[64 * 64];   // [t][k]
    __shared__ __align__(16) __bf16 Kts[64 * 64];  // [k][t]
    __shared__ __align__(16) __bf16 Vts[64 * 64];  // [d][t]
    __shared__ __align__(16) __bf16 Ss[64 * 64];   // scratch: V rows, then scores [s][t]
    __shared__ __align__(16) float  Ms[64 * 64];   // state M[d][k], f32

    for (int i = tid; i < 64 * 64; i += 512) Ms[i] = 0.0f;
    __syncthreads();

    for (int c = 0; c < SEQ / 64; ++c) {
        const int row0 = b * SEQ + c * 64;
        v8f yacc = zero8();
        v8f Macc = zero8();

        {   // load Q chunk (vectorized, 8 bf16 per thread)
            const int idx = tid * 8;
            const int s = idx >> 6, k = idx & 63;
            *(v8bf*)&Qs[idx] = *(const v8bf*)&Qb[(size_t)(row0 + s) * DIMV + hcol + k];
        }
        __syncthreads();

        // inter-chunk term: y += Q @ M_prev^T   (b-fragment = M rows, f32->bf16)
#pragma unroll
        for (int kb = 0; kb < 64; kb += 32) {
            v16bf a  = load_frag(&Qs[(16 * ti + ln15) * 64], kb, lane);
            v16bf bm = load_frag_f32(&Ms[(16 * tj + ln15) * 64], kb, lane);
            yacc = wmma_bf16(a, bm, yacc);
        }

        for (int bp = 0; bp < NB; ++bp) {
            const int krow0 = bp * SEQ + c * 64;
#if USE_TDM
            if (wv == 0) {   // TDM: K rows -> Ks ; V rows -> Ss (temp)
                tdm_load_2d(&Ks[0], &Kb[(size_t)krow0 * DIMV + hcol], 64, 64, DIMV);
                tdm_load_2d(&Ss[0], &Vb[(size_t)krow0 * DIMV + hcol], 64, 64, DIMV);
                __builtin_amdgcn_s_wait_tensorcnt(0);
            }
#else
            {   // fallback: vectorized K/V tile loads
                const int idx = tid * 8;
                const int t = idx >> 6, k = idx & 63;
                *(v8bf*)&Ks[idx] = *(const v8bf*)&Kb[(size_t)(krow0 + t) * DIMV + hcol + k];
                *(v8bf*)&Ss[idx] = *(const v8bf*)&Vb[(size_t)(krow0 + t) * DIMV + hcol + k];
            }
#endif
            __syncthreads();
#pragma unroll
            for (int e = 0; e < 8; ++e) {  // LDS transposes: Kts[k][t], Vts[d][t]
                const int idx = tid * 8 + e;
                const int r = idx >> 6, cc = idx & 63;
                Kts[cc * 64 + r] = Ks[idx];
                Vts[cc * 64 + r] = Ss[idx];
            }
            __syncthreads();

            // scores = Q @ K^T  (b-fragment = K rows)
            v8f sacc = zero8();
#pragma unroll
            for (int kb = 0; kb < 64; kb += 32) {
                v16bf a  = load_frag(&Qs[(16 * ti + ln15) * 64], kb, lane);
                v16bf bk = load_frag(&Ks[(16 * tj + ln15) * 64], kb, lane);
                sacc = wmma_bf16(a, bk, sacc);
            }
            {   // inclusive causal mask (t <= s), scale 1/B, store bf16 scores
                const int srow0 = 16 * ti + ((lane >> 4) << 3);
                const int tcol = 16 * tj + ln15;
#pragma unroll
                for (int r = 0; r < 8; ++r) {
                    const int s = srow0 + r;
                    const float v = (tcol <= s) ? 0.25f * sacc[r] : 0.0f;
                    Ss[s * 64 + tcol] = (__bf16)v;
                }
            }
            __syncthreads();

            // intra term: y += S @ V   (b-fragment = V^T rows = Vts)
            // memory delta: Macc += V^T @ K  (A = Vts rows, b-fragment = K^T rows = Kts)
#pragma unroll
            for (int kb = 0; kb < 64; kb += 32) {
                v16bf as = load_frag(&Ss[(16 * ti + ln15) * 64], kb, lane);
                v16bf bv = load_frag(&Vts[(16 * tj + ln15) * 64], kb, lane);
                yacc = wmma_bf16(as, bv, yacc);
                v16bf av  = load_frag(&Vts[(16 * ti + ln15) * 64], kb, lane);
                v16bf bkt = load_frag(&Kts[(16 * tj + ln15) * 64], kb, lane);
                Macc = wmma_bf16(av, bkt, Macc);
            }
            __syncthreads();   // before next bp overwrites Ks/Ss/Kts/Vts
        }

        {   // commit state update (disjoint tiles per wave) + write y chunk
            const int r0 = 16 * ti + ((lane >> 4) << 3);
            const int ccol = 16 * tj + ln15;
#pragma unroll
            for (int r = 0; r < 8; ++r) {
                Ms[(r0 + r) * 64 + ccol] += 0.25f * Macc[r];
                Yb[(size_t)(row0 + r0 + r) * DIMV + hcol + ccol] = (__bf16)yacc[r];
            }
        }
        __syncthreads();   // next chunk's inter-term reads updated Ms
    }
}

// ---------------------------------------------------------------------------
// Host-side orchestration
// ---------------------------------------------------------------------------
extern "C" void kernel_launch(void* const* d_in, const int* in_sizes, int n_in,
                              void* d_out, int out_size, void* d_ws, size_t ws_size,
                              hipStream_t stream) {
    (void)in_sizes; (void)n_in; (void)out_size; (void)ws_size;

    const float* x      = (const float*)d_in[0];
    const float* Wq     = (const float*)d_in[1];
    const float* Wk     = (const float*)d_in[2];
    const float* Wv     = (const float*)d_in[3];
    const float* Wo     = (const float*)d_in[4];
    const float* ln1_w  = (const float*)d_in[5];
    const float* ln1_b  = (const float*)d_in[6];
    const float* ln2_w  = (const float*)d_in[7];
    const float* ln2_b  = (const float*)d_in[8];
    const float* cmsW1  = (const float*)d_in[9];
    const float* cmsb1  = (const float*)d_in[10];
    const float* cmsW2  = (const float*)d_in[11];
    const float* cmsb2  = (const float*)d_in[12];
    float* out = (float*)d_out;

    // workspace carving (all sizes 256B-aligned)
    const size_t SZ_W   = (size_t)DIMV * DIMV * sizeof(__bf16);   // 512 KB
    const size_t SZ_ACT = (size_t)NTOK * DIMV * sizeof(__bf16);   // 8 MB
    const size_t SZ_HID = (size_t)NTOK * HIDV * sizeof(__bf16);   // 32 MB
    const size_t SZ_WT  = (size_t)HIDV * DIMV * sizeof(__bf16);   // 2 MB per level

    char* p = (char*)d_ws;
    __bf16* wqb  = (__bf16*)p; p += SZ_W;
    __bf16* wkb  = (__bf16*)p; p += SZ_W;
    __bf16* wvb  = (__bf16*)p; p += SZ_W;
    __bf16* wob  = (__bf16*)p; p += SZ_W;
    __bf16* w1t  = (__bf16*)p; p += 3 * SZ_WT;   // [l][HID][DIM]
    __bf16* w2t  = (__bf16*)p; p += 3 * SZ_WT;   // [l][DIM][HID]
    __bf16* hb   = (__bf16*)p; p += SZ_ACT;
    __bf16* qbuf = (__bf16*)p; p += SZ_ACT;
    __bf16* kbuf = (__bf16*)p; p += SZ_ACT;
    __bf16* vbuf = (__bf16*)p; p += SZ_ACT;
    __bf16* ybuf = (__bf16*)p; p += SZ_ACT;
    __bf16* hidb = (__bf16*)p; p += SZ_HID;

    // 1) weight prep
    {
        const int nw = DIMV * DIMV;
        dim3 g((nw + 255) / 256);
        cvt_bf16_kernel<<<g, 256, 0, stream>>>(Wq, wqb, nw);
        cvt_bf16_kernel<<<g, 256, 0, stream>>>(Wk, wkb, nw);
        cvt_bf16_kernel<<<g, 256, 0, stream>>>(Wv, wvb, nw);
        cvt_bf16_kernel<<<g, 256, 0, stream>>>(Wo, wob, nw);
        const int nt = DIMV * HIDV;
        dim3 gt((nt + 255) / 256);
        for (int l = 0; l < 3; ++l) {
            transpose_cvt_kernel<<<gt, 256, 0, stream>>>(cmsW1 + (size_t)l * DIMV * HIDV,
                                                         w1t + (size_t)l * HIDV * DIMV,
                                                         DIMV, HIDV);
            transpose_cvt_kernel<<<gt, 256, 0, stream>>>(cmsW2 + (size_t)l * HIDV * DIMV,
                                                         w2t + (size_t)l * DIMV * HIDV,
                                                         HIDV, DIMV);
        }
    }

    const dim3 gemm_blk(256);
    const dim3 grid_512(DIMV / 64, NTOK / 128);   // N=512
    const dim3 grid_2048(HIDV / 64, NTOK / 128);  // N=2048

    // 2) LN1 -> hb (bf16)
    ln_bf16_kernel<<<NTOK, 256, 0, stream>>>(x, ln1_w, ln1_b, hb);

    // 3) Q/K/V projections (no bias)
    gemm_wmma_kernel<512, 512, false, false, false, true>
        <<<grid_512, gemm_blk, 0, stream>>>(hb, wqb, nullptr, nullptr, qbuf);
    gemm_wmma_kernel<512, 512, false, false, false, true>
        <<<grid_512, gemm_blk, 0, stream>>>(hb, wkb, nullptr, nullptr, kbuf);
    gemm_wmma_kernel<512, 512, false, false, false, true>
        <<<grid_512, gemm_blk, 0, stream>>>(hb, wvb, nullptr, nullptr, vbuf);

    // 4) attention memory scan -> ybuf
    attn_mem_kernel<<<NB * NH, 512, 0, stream>>>(qbuf, kbuf, vbuf, ybuf);

    // 5) output projection + residual(x) -> d_out (f32)
    gemm_wmma_kernel<512, 512, false, false, true, false>
        <<<grid_512, gemm_blk, 0, stream>>>(ybuf, wob, nullptr, x, out);

    // 6) LN2 -> hb (bf16)
    ln_bf16_kernel<<<NTOK, 256, 0, stream>>>(out, ln2_w, ln2_b, hb);

    // 7) CMS chain
    for (int l = 0; l < 3; ++l) {
        gemm_wmma_kernel<512, 2048, true, true, false, true>
            <<<grid_2048, gemm_blk, 0, stream>>>(hb, w1t + (size_t)l * HIDV * DIMV,
                                                 cmsb1 + (size_t)l * HIDV, nullptr, hidb);
        if (l < 2) {
            gemm_wmma_kernel<2048, 512, true, false, false, true>
                <<<grid_512, gemm_blk, 0, stream>>>(hidb, w2t + (size_t)l * DIMV * HIDV,
                                                    cmsb2 + (size_t)l * DIMV, nullptr, hb);
        } else {
            gemm_wmma_kernel<2048, 512, true, false, true, false>
                <<<grid_512, gemm_blk, 0, stream>>>(hidb, w2t + (size_t)l * DIMV * HIDV,
                                                    cmsb2 + (size_t)l * DIMV, out, out);
        }
    }
}
